// Correlation_69114613728076
// MI455X (gfx1250) — matrix operand order repaired and checked by reading, not compiled, not run on hardware
//
#include <hip/hip_runtime.h>

// Correlation / cost-volume (FlowNet): out[b, 9*i+j, h, w] =
//   (1/64) * sum_c x1[b,c,h,w] * x2[b,c,h+i-4,w+j-4]   (zero padded)
//
// MI455X (gfx1250, wave32):
//  - memory-bound target (~247 MB one-pass @ 23.3 TB/s ~= 10.6 us); x1+x2
//    (151 MB) fit in 192 MB L2 -> x2 temporal (halo reused by neighboring-h
//    blocks), x1 nontemporal load (zero reuse), output nontemporal store.
//  - channel dots as f16 WMMA: D[m,n] = sum_c x1[c,w0+m]*x2[c,w0-4+n]; the
//    useful outputs are the diagonal band j=n-m in [0,8]. 2 N-tiles x 2
//    K-steps x 9 di x 2 h-rows = 72 v_wmma_f32_16x16x32_f16 per block (8 per wave).
//  - 1/64 mean is folded into the x1 f16 conversion (exact: power of two).
//  - C tiles are stored to LDS unconditionally (8 immediate-offset
//    ds_store_b32 per wave-tile, no exec masking); the diagonal band is
//    extracted during the drain and written as coalesced 128-bit NT stores.

#define BATCH 8
#define CCH   64
#define HH    192
#define WW    192
#define MD    4
#define OD    9            // 2*MD+1
#define ND    81           // OD*OD
#define WT    16           // output w-tile per block
#define HT    2            // output h-rows per block
#define RB    (HT + 2*MD)  // 10 staged x2 rows
#define NBW   32           // B-strip row pitch in n (24 columns staged)
#define RS    80           // f16 LDS row stride (64 ch + pad; 160B, 32B-aligned)
#define NS    33           // C-tile row pitch in floats (16..32 used, +1 pad)
#define TP    (WT * NS)    // C-tile size: 528 floats
#define NTHREADS 288       // 9 wave32s, wave == di

typedef __attribute__((ext_vector_type(16))) _Float16 v16h;
typedef __attribute__((ext_vector_type(8)))  _Float16 v8h;
typedef __attribute__((ext_vector_type(8)))  float    v8f;
typedef __attribute__((ext_vector_type(4)))  float    v4f;

__global__ __launch_bounds__(NTHREADS)
void corr81_wmma_f16(const float* __restrict__ x1,
                     const float* __restrict__ x2,
                     float* __restrict__ out) {
  __shared__ __align__(32) _Float16 lA[HT * WT * RS];    //  5 KB: x1 tiles [hh][w][c]
  __shared__ __align__(32) _Float16 lB[RB * NBW * RS];   // 50 KB: x2 halo [rr][n][c]
  __shared__ __align__(16) float    lC[HT * OD * TP];    // 38 KB: C tiles [hh][di][m][n]

  const int w0  = blockIdx.x * WT;
  const int h0  = blockIdx.y * HT;
  const int b   = blockIdx.z;
  const int tid = threadIdx.x;

  // ---- stage f32 global -> f16 LDS in float4 quads along w ----
  const int QA = HT * CCH * (WT / 4);   // 512 x1 quads
  const int QB = RB * CCH * 6;          // 3840 x2 quads (24 cols = 6 quads/row)
  for (int q = tid; q < QA + QB; q += NTHREADS) {
    if (q < QA) {
      const int wq = (q & 3) * 4;
      const int c  = (q >> 2) & (CCH - 1);
      const int hh = q >> 8;
      const v4f v = __builtin_nontemporal_load(reinterpret_cast<const v4f*>(
          x1 + (((size_t)b * CCH + c) * HH + (h0 + hh)) * WW + w0 + wq));
      const int base = (hh * WT + wq) * RS + c;
      // fold the channel mean (1/64 = 2^-6, exact in f16) into x1
      lA[base + 0 * RS] = (_Float16)(v.x * 0.015625f);
      lA[base + 1 * RS] = (_Float16)(v.y * 0.015625f);
      lA[base + 2 * RS] = (_Float16)(v.z * 0.015625f);
      lA[base + 3 * RS] = (_Float16)(v.w * 0.015625f);
    } else {
      const int qb  = q - QA;
      const int nq  = qb % 6;
      const int tmp = qb / 6;
      const int c   = tmp & (CCH - 1);
      const int rr  = tmp >> 6;             // staged x2 row 0..9
      const int h2  = h0 + rr - MD;
      const int w2q = w0 - MD + nq * 4;     // quad fully in or fully out of [0,W)
      v4f v = {0.f, 0.f, 0.f, 0.f};
      if ((unsigned)h2 < (unsigned)HH && (unsigned)w2q < (unsigned)WW)
        v = *reinterpret_cast<const v4f*>(
            x2 + (((size_t)b * CCH + c) * HH + h2) * WW + w2q);   // temporal: L2 reuse
      const int base = (rr * NBW + nq * 4) * RS + c;
      lB[base + 0 * RS] = (_Float16)v.x;
      lB[base + 1 * RS] = (_Float16)v.y;
      lB[base + 2 * RS] = (_Float16)v.z;
      lB[base + 3 * RS] = (_Float16)v.w;
    }
  }
  __syncthreads();

  const int lane = tid & 31;
  const int wv   = tid >> 5;        // wave index == di (0..8)
  const int row  = lane & 15;
  const int hi   = lane >> 4;
  const int kb   = hi * 8;          // unused; A frag uses kb2: lanes 0-15 K 0..15, 16-31 K 16..31
  const int kb2  = hi * 16;         // B frag: lanes 0-15 K 0..15, 16-31 K 16..31

  #pragma unroll
  for (int hh = 0; hh < HT; ++hh) {
    const _Float16* arow = &lA[(hh * WT + row) * RS];
    const v8h alo0 = *(const v8h*)&arow[kb2];
    const v8h ahi0 = *(const v8h*)&arow[8 + kb2];
    const v8h alo1 = *(const v8h*)&arow[32 + kb2];
    const v8h ahi1 = *(const v8h*)&arow[40 + kb2];
    const v16h a0 = __builtin_shufflevector(alo0, ahi0, 0,1,2,3,4,5,6,7,8,9,10,11,12,13,14,15);
    const v16h a1 = __builtin_shufflevector(alo1, ahi1, 0,1,2,3,4,5,6,7,8,9,10,11,12,13,14,15);
    const int rr = hh + wv;                 // staged x2 row for this (hh, di)
    float* ctile = &lC[(hh * OD + wv) * TP];

    #pragma unroll
    for (int t = 0; t < 2; ++t) {
      const int n = t * 16 + row;           // strip column: x2 w = w0 - 4 + n
      const _Float16* brow = &lB[(rr * NBW + n) * RS];
      const v16h b0 = *(const v16h*)&brow[kb2];        // K 0..31
      const v16h b1 = *(const v16h*)&brow[32 + kb2];   // K 32..63

      v8f acc = {};
      acc = __builtin_amdgcn_wmma_f32_16x16x32_f16(false, a0, false, b0,
                                                   (short)0, acc, false, false);
      acc = __builtin_amdgcn_wmma_f32_16x16x32_f16(false, a1, false, b1,
                                                   (short)0, acc, false, false);

      // C/D layout: lane holds column N=row of this tile, VGPR r -> M=r+8*hi.
      // Unconditional tile store: one base address, 8 immediate-offset stores.
      float* cdst = ctile + (hi * 8) * NS + n;
      #pragma unroll
      for (int r = 0; r < 8; ++r)
        cdst[r * NS] = acc[r];
    }
  }
  __syncthreads();

  // ---- drain: band extract + coalesced 128-bit NT stores (64B / 4 lanes) ----
  for (int qq = tid; qq < HT * ND * 4; qq += NTHREADS) {   // 648 quads
    const int hh = qq / (ND * 4);
    const int r2 = qq - hh * (ND * 4);
    const int d  = r2 >> 2;                 // displacement channel 0..80
    const int mq = (r2 & 3) * 4;            // quad start in m
    const int di = d / OD;
    const int j  = d - di * OD;
    const float* ct = &lC[(hh * OD + di) * TP + mq * (NS + 1) + j];  // C[m][m+j]
    v4f val;
    val.x = ct[0 * (NS + 1)];
    val.y = ct[1 * (NS + 1)];
    val.z = ct[2 * (NS + 1)];
    val.w = ct[3 * (NS + 1)];
    float* dst = out + (((size_t)b * ND + d) * HH + (h0 + hh)) * WW + w0 + mq;
    __builtin_nontemporal_store(val, reinterpret_cast<v4f*>(dst));
  }
}

extern "C" void kernel_launch(void* const* d_in, const int* in_sizes, int n_in,
                              void* d_out, int out_size, void* d_ws, size_t ws_size,
                              hipStream_t stream) {
  (void)in_sizes; (void)n_in; (void)d_ws; (void)ws_size; (void)out_size;
  const float* x1 = (const float*)d_in[0];
  const float* x2 = (const float*)d_in[1];
  float* out = (float*)d_out;

  dim3 grid(WW / WT, HH / HT, BATCH);   // 12 x 96 x 8 = 9216 blocks
  corr81_wmma_f16<<<grid, NTHREADS, 0, stream>>>(x1, x2, out);
}
